// MixtureOfExperts_9569187135823
// MI455X (gfx1250) — matrix operand order repaired and checked by reading, not compiled, not run on hardware
//
#include <hip/hip_runtime.h>
#include <hip/hip_bf16.h>

#define DD 1024   // input dim
#define HH 4096   // expert hidden dim
#define EE 8      // experts
#define TT 4096   // tokens

#define MTOK 32             // tokens per block (2 WMMA N-groups of 16)
#define TSTR 40             // LDS token-dim stride (mult of 8 -> 16B aligned b128 ops)
#define HSLICE 512          // H processed in 8 slices of 512 columns

typedef __attribute__((ext_vector_type(16))) __bf16 v16bf;
typedef __attribute__((ext_vector_type(8)))  __bf16 v8bf;
typedef __attribute__((ext_vector_type(8)))  float  v8f;

// ---------------- kernel 0: zero output + bucket counts ----------------
__global__ void moe_zero(float* __restrict__ out, int* __restrict__ counts) {
    if (blockIdx.x == 0 && threadIdx.x < EE) counts[threadIdx.x] = 0;
    size_t stride = (size_t)gridDim.x * blockDim.x;
    for (size_t i = (size_t)blockIdx.x * blockDim.x + threadIdx.x;
         i < (size_t)TT * DD; i += stride)
        out[i] = 0.0f;
}

// ---------------- kernel 1: routing (one wave32 per token) ----------------
__global__ void moe_router(const float* __restrict__ x, const float* __restrict__ Wc,
                           const float* __restrict__ bc, int* __restrict__ counts,
                           int* __restrict__ tidList, float* __restrict__ wList) {
    int wv   = threadIdx.x >> 5;
    int lane = threadIdx.x & 31;
    int t    = blockIdx.x * 8 + wv;           // grid = TT/8 blocks of 256
    const float* xr = x + (size_t)t * DD;

    float acc[EE];
#pragma unroll
    for (int e = 0; e < EE; ++e) acc[e] = 0.0f;
    for (int d = lane; d < DD; d += 32) {
        float xv = xr[d];
#pragma unroll
        for (int e = 0; e < EE; ++e) acc[e] = fmaf(xv, Wc[e * DD + d], acc[e]);
    }
#pragma unroll
    for (int e = 0; e < EE; ++e)
        for (int off = 16; off > 0; off >>= 1)
            acc[e] += __shfl_xor(acc[e], off, 32);

    if (lane == 0) {
        float lg[EE], p[EE];
        float mx = -3.4e38f;
#pragma unroll
        for (int e = 0; e < EE; ++e) { lg[e] = acc[e] + bc[e]; mx = fmaxf(mx, lg[e]); }
        float sum = 0.0f;
#pragma unroll
        for (int e = 0; e < EE; ++e) { p[e] = __expf(lg[e] - mx); sum += p[e]; }
        float inv = 1.0f / sum;
#pragma unroll
        for (int e = 0; e < EE; ++e) p[e] *= inv;   // softmax over ALL experts (no top-k renorm)
        int i0 = 0;
#pragma unroll
        for (int e = 1; e < EE; ++e) if (p[e] > p[i0]) i0 = e;     // ties -> lowest index
        int i1 = (i0 == 0) ? 1 : 0;
#pragma unroll
        for (int e = 0; e < EE; ++e) if (e != i0 && p[e] > p[i1]) i1 = e;

        int s0 = atomicAdd(&counts[i0], 1);
        tidList[i0 * TT + s0] = t;  wList[i0 * TT + s0] = p[i0];
        int s1 = atomicAdd(&counts[i1], 1);
        tidList[i1 * TT + s1] = t;  wList[i1 * TT + s1] = p[i1];
    }
}

// ---------------- WMMA fragment loaders (layouts per CDNA5 ISA 7.12.2) ----------------
// A (16x32 bf16) from GLOBAL fp32 W (row-major, rowStride floats):
// lane m%16 holds row (rowTile + m); elems 0..7 = K[k0+off..], 8..15 = K[k0+16+off..]
// -> 4x global_load_b128 per fragment, 32B-aligned.
__device__ __forceinline__ v16bf ldA_gl(const float* __restrict__ Wbase, int rowStride,
                                        int rowTile, int lane, int k0) {
    const float* p = Wbase + (size_t)(rowTile + (lane & 15)) * rowStride
                   + k0 + ((lane & 16) ? 8 : 0);
    float4 q0 = *reinterpret_cast<const float4*>(p);
    float4 q1 = *reinterpret_cast<const float4*>(p + 4);
    float4 q2 = *reinterpret_cast<const float4*>(p + 16);
    float4 q3 = *reinterpret_cast<const float4*>(p + 20);
    v16bf a;
    a[0]=(__bf16)q0.x; a[1]=(__bf16)q0.y; a[2]=(__bf16)q0.z; a[3]=(__bf16)q0.w;
    a[4]=(__bf16)q1.x; a[5]=(__bf16)q1.y; a[6]=(__bf16)q1.z; a[7]=(__bf16)q1.w;
    a[8]=(__bf16)q2.x; a[9]=(__bf16)q2.y; a[10]=(__bf16)q2.z; a[11]=(__bf16)q2.w;
    a[12]=(__bf16)q3.x; a[13]=(__bf16)q3.y; a[14]=(__bf16)q3.z; a[15]=(__bf16)q3.w;
    return a;
}

// B (32x16 bf16) from LDS transposed tile T[k][token] (stride TSTR):
// lane L holds token t0+L%16; elems 0..7 = K[k0+off..], 8..15 = K[k0+16+off..] (off = 8*(L/16)).
__device__ __forceinline__ v16bf ldB_lds(const __bf16* __restrict__ base, int lane,
                                         int k0, int t0) {
    const __bf16* p = base + (size_t)(k0 + ((lane & 16) ? 8 : 0)) * TSTR + t0 + (lane & 15);
    v16bf b;
#pragma unroll
    for (int s = 0; s < 8; ++s) { b[s] = p[s * TSTR]; b[8 + s] = p[(16 + s) * TSTR]; }
    return b;
}

// ---------------- kernel 2: fused grouped expert FFN (bf16 WMMA, W-as-A) ----------------
// grid = EE * (TT/MTOK): blockIdx = e*128 + tile -> tiles of one expert contiguous (L2 reuse)
__global__ __launch_bounds__(256) void moe_ffn(
    const float* __restrict__ x, const float* __restrict__ W1, const float* __restrict__ b1,
    const float* __restrict__ W2, const float* __restrict__ b2,
    const int* __restrict__ counts, const int* __restrict__ tidList,
    const float* __restrict__ wList, float* __restrict__ out) {

    int e    = blockIdx.x >> 7;
    int tile = blockIdx.x & 127;
    int cnt  = counts[e];
    if (tile * MTOK >= cnt) return;   // uniform across block: no divergent barriers

    extern __shared__ __align__(16) char smem[];
    __bf16* Xt   = (__bf16*)smem;                                   // [DD][TSTR]  x^T tile
    __bf16* Ht   = (__bf16*)(smem + (size_t)DD * TSTR * 2);         // [HSLICE][TSTR] h^T slice
    int*    sTid = (int*)  (smem + (size_t)(DD + HSLICE) * TSTR * 2);
    float*  sW   = (float*)(smem + (size_t)(DD + HSLICE) * TSTR * 2 + 128);

    if (threadIdx.x < MTOK) {
        int slot = tile * MTOK + threadIdx.x;
        if (slot < cnt) { sTid[threadIdx.x] = tidList[e * TT + slot];
                          sW[threadIdx.x]   = wList[e * TT + slot]; }
        else            { sTid[threadIdx.x] = -1; sW[threadIdx.x] = 0.0f; }
    }
    __syncthreads();

    // Stage X tile transposed into LDS as bf16: Xt[d][m] (padded rows -> 0)
    for (int idx = threadIdx.x; idx < MTOK * DD; idx += 256) {
        int m = idx >> 10;          // token-in-tile
        int d = idx & (DD - 1);     // contiguous in d -> coalesced global read
        int t = sTid[m];
        float v = (t >= 0) ? x[(size_t)t * DD + d] : 0.0f;
        Xt[d * TSTR + m] = (__bf16)v;
    }
    __syncthreads();

    const float* W1e = W1 + (size_t)e * HH * DD;
    const float* W2e = W2 + (size_t)e * DD * HH;
    const float* b1e = b1 + (size_t)e * HH;
    const float* b2e = b2 + (size_t)e * DD;

    int wv    = threadIdx.x >> 5;
    int lane  = threadIdx.x & 31;
    int n     = lane & 15;
    int mbase = (lane & 16) ? 8 : 0;

    v8f yAcc[8][2];                 // wave owns d-rows [wv*128, wv*128+128): 8 tiles x 2 groups
#pragma unroll
    for (int dt = 0; dt < 8; ++dt) { yAcc[dt][0] = {}; yAcc[dt][1] = {}; }

    for (int oct = 0; oct < HH / HSLICE; ++oct) {   // 8 slices of 512 H-columns
        int hbase = oct * HSLICE;

        // ---- Phase B: Ht[0..511][token] = relu(W1_slice @ x^T + b1)^T (bf16)
        for (int pass = 0; pass < HSLICE / 128; ++pass) {   // 4 passes x (8 waves * 16 rows)
            int hrowTile = hbase + pass * 128 + wv * 16;
            v8f acc0 = {}, acc1 = {};
            for (int k0 = 0; k0 < DD; k0 += 32) {
                v16bf a  = ldA_gl(W1e, DD, hrowTile, lane, k0);
                v16bf b0 = ldB_lds(Xt, lane, k0, 0);
                v16bf b1v = ldB_lds(Xt, lane, k0, 16);
                acc0 = __builtin_amdgcn_wmma_f32_16x16x32_bf16(
                           false, a, false, b0, (short)0, acc0, false, false);
                acc1 = __builtin_amdgcn_wmma_f32_16x16x32_bf16(
                           false, a, false, b1v, (short)0, acc1, false, false);
            }
#pragma unroll
            for (int r = 0; r < 8; ++r) {       // C elem r -> (m = h-row mbase+r, n = token)
                int hrow  = hrowTile + mbase + r;
                float bias = b1e[hrow];
                float v0 = acc0[r] + bias; v0 = v0 > 0.0f ? v0 : 0.0f;
                float v1 = acc1[r] + bias; v1 = v1 > 0.0f ? v1 : 0.0f;
                Ht[(hrow - hbase) * TSTR + n]      = (__bf16)v0;
                Ht[(hrow - hbase) * TSTR + 16 + n] = (__bf16)v1;
            }
        }
        __syncthreads();

        // ---- Phase C: yAcc += W2[:, hbase:hbase+512] @ h^T_slice
#pragma unroll
        for (int dt = 0; dt < 8; ++dt) {
            int drowTile = wv * 128 + dt * 16;
            v8f a0 = yAcc[dt][0], a1 = yAcc[dt][1];
            for (int k0 = 0; k0 < HSLICE; k0 += 32) {
                v16bf a  = ldA_gl(W2e, HH, drowTile, lane, hbase + k0);
                v16bf b0 = ldB_lds(Ht, lane, k0, 0);
                v16bf b1v = ldB_lds(Ht, lane, k0, 16);
                a0 = __builtin_amdgcn_wmma_f32_16x16x32_bf16(
                         false, a, false, b0, (short)0, a0, false, false);
                a1 = __builtin_amdgcn_wmma_f32_16x16x32_bf16(
                         false, a, false, b1v, (short)0, a1, false, false);
            }
            yAcc[dt][0] = a0; yAcc[dt][1] = a1;
        }
        __syncthreads();   // before next slice overwrites Ht
    }

    // ---- Epilogue: out[t][d] += s_t * (y + b2); exactly 2 commutative adds per element
#pragma unroll
    for (int dt = 0; dt < 8; ++dt) {
        int drowTile = wv * 128 + dt * 16;
#pragma unroll
        for (int r = 0; r < 8; ++r) {
            int drow = drowTile + mbase + r;
            float bias2 = b2e[drow];
#pragma unroll
            for (int g = 0; g < 2; ++g) {
                int m = g * 16 + n;           // token slot in tile
                int t = sTid[m];
                float s = sW[m];              // 0.0 for padded rows -> adds exact 0
                if (t < 0) t = 0;
                atomicAdd(&out[(size_t)t * DD + drow], (yAcc[dt][g][r] + bias2) * s);
            }
        }
    }
}

// ---------------- host launcher ----------------
extern "C" void kernel_launch(void* const* d_in, const int* in_sizes, int n_in,
                              void* d_out, int out_size, void* d_ws, size_t ws_size,
                              hipStream_t stream) {
    const float* x  = (const float*)d_in[0];
    const float* W1 = (const float*)d_in[1];
    const float* b1 = (const float*)d_in[2];
    const float* W2 = (const float*)d_in[3];
    const float* b2 = (const float*)d_in[4];
    const float* Wc = (const float*)d_in[5];
    const float* bc = (const float*)d_in[6];
    float* out = (float*)d_out;

    // workspace: [counts: 256B] [tidList: E*T ints] [wList: E*T floats]
    int*   counts  = (int*)d_ws;
    int*   tidList = (int*)((char*)d_ws + 256);
    float* wList   = (float*)((char*)d_ws + 256 + (size_t)EE * TT * sizeof(int));

    moe_zero  <<<1024,   256, 0, stream>>>(out, counts);
    moe_router<<<TT / 8, 256, 0, stream>>>(x, Wc, bc, counts, tidList, wList);

    size_t smemBytes = (size_t)(DD + HSLICE) * TSTR * 2   // Xt + Ht (bf16)
                     + 256;                               // token ids + scores
    moe_ffn<<<EE * (TT / MTOK), 256, smemBytes, stream>>>(
        x, W1, b1, W2, b2, counts, tidList, wList, out);
}
